// RITS_75084618269438
// MI455X (gfx1250) — compile-verified, matrix operand order checked by reading, not run on hardware
//
#include <hip/hip_runtime.h>

// ---------------- CDNA5 WMMA types ----------------
typedef _Float16 v16h __attribute__((ext_vector_type(16)));
typedef float    v8f  __attribute__((ext_vector_type(8)));

#define WMMA_F16(a, b, c) \
  __builtin_amdgcn_wmma_f32_16x16x32_f16(false, (a), false, (b), (short)0, (c), false, false)

constexpr int B_ = 256, T_ = 128, F_ = 128, H_ = 512;
constexpr int BT = 16;          // batch rows per workgroup (one WMMA M tile)
constexpr int NTHREADS = 256;   // 8 wave32 -> one WGP

union AFrag { v16h h; unsigned u[8]; };
union CFrag { v8f  v; float    f[8]; };

// Launder a pointer through an empty asm so loads derived from it cannot be
// hoisted out of the time loop (prevents LICM -> register blowup -> scratch spill).
template <typename T>
__device__ __forceinline__ T* opaque(const T* p) {
  unsigned long long v = (unsigned long long)p;
  asm volatile("" : "+s"(v));
  return (T*)v;
}

// A-fragment (16x32 f16, M on lanes) from row-major f16 LDS [16][K]
__device__ __forceinline__ v16h load_a(const _Float16* __restrict a, int K, int kbase, int lane) {
  int m = lane & 15, hf = lane >> 4;
  AFrag r;
#pragma unroll
  for (int j = 0; j < 8; ++j) {
    int k0 = (j < 4) ? (8 * hf + 2 * j) : (16 + 8 * hf + 2 * (j - 4));
    r.u[j] = *(const unsigned*)(a + m * K + kbase + k0);   // pair (k,k+1), dword aligned
  }
  return r.h;
}

// B-fragment (32x16 f16, N on lanes) from fragment-major packed weights:
// each fragment is 1KB contiguous; lane l owns dwords [l*8, l*8+8)
__device__ __forceinline__ v16h load_b(const uint4* __restrict pk, int frag, int lane) {
  union { v16h h; uint4 u[2]; } r;
  const uint4* p = pk + (size_t)frag * 64 + lane * 2;
  r.u[0] = p[0];
  r.u[1] = p[1];
  return r.h;
}

__device__ __forceinline__ float sigm(float x) { return 1.0f / (1.0f + __expf(-x)); }

// ---------------- weight packer: W[O][I] f32 -> fragment-major f16 ----------------
// B[k][n] = W[n_global][k_global]; fragment index = ntile*KC + kchunk
__global__ void pack_w(const float* __restrict W, _Float16* __restrict dst,
                       int O, int I, int zero_diag) {
  int KC = I >> 5;
  int total = (O >> 4) * KC * 256;     // dwords
  int tid = blockIdx.x * 256 + threadIdx.x;
  if (tid >= total) return;
  int bi = tid >> 8, s = tid & 255;
  int l = s >> 3, j = s & 7;
  int nt = bi / KC, kc = bi - nt * KC;
  int n = l & 15, hf = l >> 4;
  int k0 = (j < 4) ? (8 * hf + 2 * j) : (16 + 8 * hf + 2 * (j - 4));
  int o = nt * 16 + n;
  int k = kc * 32 + k0;
  float w0 = W[(size_t)o * I + k];
  float w1 = W[(size_t)o * I + k + 1];
  if (zero_diag) { if (o == k) w0 = 0.f; if (o == k + 1) w1 = 0.f; }
  dst[(size_t)tid * 2]     = (_Float16)w0;
  dst[(size_t)tid * 2 + 1] = (_Float16)w1;
}

// ------- aux: td_x diagonal, per-t denom, combined gate bias, zero loss slot -------
__global__ void prep_aux(const float* __restrict td_x_W, const float* __restrict tmask,
                         const float* __restrict bih, const float* __restrict bhh,
                         float* __restrict diag, float* __restrict denom,
                         float* __restrict gbias, float* __restrict loss_slot) {
  __shared__ float red[256];
  int t = blockIdx.x, tid = threadIdx.x;
  if (t == 0 && tid < F_) diag[tid] = td_x_W[tid * F_ + tid];
  if (t == 0 && tid == 0) *loss_slot = 0.f;
  if (t == 1) {
    for (int i = tid; i < 4 * H_; i += 256) gbias[i] = bih[i] + bhh[i];
  }
  float s = 0.f;
  for (int i = tid; i < B_ * F_; i += 256) {
    int b = i >> 7, f = i & (F_ - 1);
    s += tmask[((size_t)b * T_ + t) * F_ + f];
  }
  red[tid] = s; __syncthreads();
  for (int k = 128; k > 0; k >>= 1) { if (tid < k) red[tid] += red[tid + k]; __syncthreads(); }
  if (tid == 0) denom[t] = red[0] + 1e-8f;
}

// ---------------- main persistent recurrence kernel ----------------
__global__ void __launch_bounds__(NTHREADS, 1)
rits_main(const float* __restrict x, const float* __restrict mI, const float* __restrict dI,
          const float* __restrict tgt, const float* __restrict ev,
          const float* __restrict td_h_b, const float* __restrict td_x_b,
          const float* __restrict hist_b, const float* __restrict feat_b,
          const float* __restrict wc_b, const float* __restrict gbias,
          const uint4* __restrict tdhP, const uint4* __restrict histP,
          const uint4* __restrict featP, const uint4* __restrict wcP,
          const uint4* __restrict ihP, const uint4* __restrict hhP,
          const float* __restrict diag, const float* __restrict denom,
          float* __restrict out) {
  extern __shared__ char smem[];
  float*    h32  = (float*)(smem + 0);        // [16][512] f32   32KB
  float*    c32  = (float*)(smem + 32768);    // [16][512] f32   32KB
  _Float16* h16  = (_Float16*)(smem + 65536); // [16][512] f16   16KB (post-decay h)
  float*    x32  = (float*)(smem + 81920);    // [16][128] f32
  float*    tg32 = (float*)(smem + 90112);    // [16][128] f32
  float*    xh32 = (float*)(smem + 98304);    // [16][128] f32  x_hist
  _Float16* d16  = (_Float16*)(smem + 106496);// [16][128] f16
  _Float16* m16  = (_Float16*)(smem + 110592);// [16][128] f16
  _Float16* ev16 = (_Float16*)(smem + 114688);// [16][128] f16
  _Float16* xc16 = (_Float16*)(smem + 118784);// [16][128] f16
  _Float16* cat1 = (_Float16*)(smem + 122880);// [16][256] f16  [gamma_x | m]
  _Float16* cat2 = (_Float16*)(smem + 131072);// [16][256] f16  [c_c | m]
  float*    red  = (float*)(smem + 139264);   // [256] f32

  const int tid  = threadIdx.x;
  const int lane = tid & 31, w = tid >> 5;
  const int hf = lane >> 4, n = lane & 15;
  const int b0 = blockIdx.x * BT;

  for (int i = tid; i < BT * H_; i += NTHREADS) { h32[i] = 0.f; c32[i] = 0.f; h16[i] = (_Float16)0.f; }
  __syncthreads();

  float lacc = 0.f;

  for (int t = 0; t < T_; ++t) {
    const float invden = 1.0f / denom[t];

    // -- phase 1: stage inputs, gamma_x (diag path), cat halves --
    {
      const float* diag_l = opaque(diag);
      const float* txb_l  = opaque(td_x_b);
      for (int i = tid; i < BT * F_; i += NTHREADS) {
        int r = i >> 7, f = i & (F_ - 1);
        size_t g = ((size_t)(b0 + r) * T_ + t) * F_ + f;
        float xv = x[g], mv = mI[g], dv = dI[g], tv = tgt[g], e = ev[g];
        x32[i] = xv; tg32[i] = tv;
        m16[i] = (_Float16)mv; d16[i] = (_Float16)dv; ev16[i] = (_Float16)e;
        float gx = __expf(-fmaxf(dv * diag_l[f] + txb_l[f], 0.f));
        cat1[r * (2 * F_) + f]      = (_Float16)gx;
        cat1[r * (2 * F_) + F_ + f] = (_Float16)mv;
        cat2[r * (2 * F_) + F_ + f] = (_Float16)mv;
      }
    }
    __syncthreads();

    // -- phase 2: gamma_h = exp(-relu(d @ tdh^T + b)); h *= gamma_h; refresh h16 --
    for (int q = 0; q < 4; ++q) {
      const uint4* tdhP_l = opaque(tdhP);
      const float* thb_l  = opaque(td_h_b);
      int nt = w + 8 * q;                       // 32 H-column tiles
      CFrag c;
#pragma unroll
      for (int j = 0; j < 8; ++j) c.f[j] = 0.f;
#pragma unroll
      for (int kc = 0; kc < 4; ++kc) {
        v16h a = load_a(d16, F_, kc * 32, lane);
        c.v = WMMA_F16(a, load_b(tdhP_l, nt * 4 + kc, lane), c.v);
      }
#pragma unroll
      for (int r = 0; r < 8; ++r) {
        int m = r + 8 * hf, col = nt * 16 + n;
        float g = __expf(-fmaxf(c.f[r] + thb_l[col], 0.f));
        float hv = h32[m * H_ + col] * g;
        h32[m * H_ + col] = hv;
        h16[m * H_ + col] = (_Float16)hv;
      }
    }
    __syncthreads();

    // -- phase 3: x_hist = h @ hist_W^T + b; x_c = m*x + (1-m)*x_hist --
    {
      const uint4* histP_l = opaque(histP);
      const float* hb_l    = opaque(hist_b);
      CFrag c;
#pragma unroll
      for (int j = 0; j < 8; ++j) c.f[j] = 0.f;
#pragma unroll 4
      for (int kc = 0; kc < 16; ++kc) {
        v16h a = load_a(h16, H_, kc * 32, lane);
        c.v = WMMA_F16(a, load_b(histP_l, w * 16 + kc, lane), c.v);
      }
#pragma unroll
      for (int r = 0; r < 8; ++r) {
        int m = r + 8 * hf, col = w * 16 + n;
        float xh = c.f[r] + hb_l[col];
        xh32[m * F_ + col] = xh;
        float mv = (float)m16[m * F_ + col];
        float xc = mv * x32[m * F_ + col] + (1.f - mv) * xh;
        xc16[m * F_ + col] = (_Float16)xc;
      }
    }
    __syncthreads();

    // -- phase 4: z_h, alpha, c_h, c_c (fused), imputation out, loss --
    {
      const uint4* featP_l = opaque(featP);
      const uint4* wcP_l   = opaque(wcP);
      const float* fb_l    = opaque(feat_b);
      const float* wb_l    = opaque(wc_b);
      CFrag zc, ac;
#pragma unroll
      for (int j = 0; j < 8; ++j) { zc.f[j] = 0.f; ac.f[j] = 0.f; }
#pragma unroll
      for (int kc = 0; kc < 4; ++kc) {
        v16h a = load_a(xc16, F_, kc * 32, lane);
        zc.v = WMMA_F16(a, load_b(featP_l, w * 4 + kc, lane), zc.v);
      }
#pragma unroll 4
      for (int kc = 0; kc < 8; ++kc) {
        v16h a = load_a(cat1, 2 * F_, kc * 32, lane);
        ac.v = WMMA_F16(a, load_b(wcP_l, w * 8 + kc, lane), ac.v);
      }
#pragma unroll
      for (int r = 0; r < 8; ++r) {
        int m = r + 8 * hf, col = w * 16 + n;
        float z  = zc.f[r] + fb_l[col];
        float al = sigm(ac.f[r] + wb_l[col]);
        float xh = xh32[m * F_ + col];
        float ch = al * z + (1.f - al) * xh;
        float mv = (float)m16[m * F_ + col];
        float cc = mv * x32[m * F_ + col] + (1.f - mv) * ch;
        out[((size_t)(b0 + m) * T_ + t) * F_ + col] = cc;
        cat2[m * (2 * F_) + col] = (_Float16)cc;
        float tv = tg32[m * F_ + col], e = (float)ev16[m * F_ + col];
        float d1 = xh - tv, d2 = z - tv, d3 = ch - tv;
        lacc += e * (d1 * d1 + d2 * d2 + d3 * d3) * invden;
      }
    }
    __syncthreads();

    // -- phase 5: gates = [c_c,m]@W_ih^T + h@W_hh^T (+bias), LSTM update --
    // wave owns matching (i,f,g,o) column tiles -> LSTM applied from accumulators
    for (int q = 0; q < 4; ++q) {
      const uint4* ihP_l = opaque(ihP);
      const uint4* hhP_l = opaque(hhP);
      const float* gb_l  = opaque(gbias);
      int ct = w + 8 * q;                       // 32 H-column tiles
      CFrag gi, gf, gg, go;
#pragma unroll
      for (int j = 0; j < 8; ++j) { gi.f[j] = 0.f; gf.f[j] = 0.f; gg.f[j] = 0.f; go.f[j] = 0.f; }
#pragma unroll 2
      for (int kc = 0; kc < 8; ++kc) {          // K = 2F over [c_c | m]
        v16h a = load_a(cat2, 2 * F_, kc * 32, lane);
        gi.v = WMMA_F16(a, load_b(ihP_l, (0 * 32 + ct) * 8 + kc, lane), gi.v);
        gf.v = WMMA_F16(a, load_b(ihP_l, (1 * 32 + ct) * 8 + kc, lane), gf.v);
        gg.v = WMMA_F16(a, load_b(ihP_l, (2 * 32 + ct) * 8 + kc, lane), gg.v);
        go.v = WMMA_F16(a, load_b(ihP_l, (3 * 32 + ct) * 8 + kc, lane), go.v);
      }
#pragma unroll 2
      for (int kc = 0; kc < 16; ++kc) {         // K = H over decayed h
        v16h a = load_a(h16, H_, kc * 32, lane);
        gi.v = WMMA_F16(a, load_b(hhP_l, (0 * 32 + ct) * 16 + kc, lane), gi.v);
        gf.v = WMMA_F16(a, load_b(hhP_l, (1 * 32 + ct) * 16 + kc, lane), gf.v);
        gg.v = WMMA_F16(a, load_b(hhP_l, (2 * 32 + ct) * 16 + kc, lane), gg.v);
        go.v = WMMA_F16(a, load_b(hhP_l, (3 * 32 + ct) * 16 + kc, lane), go.v);
      }
      // LSTM elementwise: writes only wave-owned (m,col) of c32/h32; h16 refreshed in phase 2
#pragma unroll
      for (int r = 0; r < 8; ++r) {
        int m = r + 8 * hf, col = ct * 16 + n;
        float ig = sigm(gi.f[r] + gb_l[0 * H_ + col]);
        float fg = sigm(gf.f[r] + gb_l[1 * H_ + col]);
        float gt = tanhf(gg.f[r] + gb_l[2 * H_ + col]);
        float og = sigm(go.f[r] + gb_l[3 * H_ + col]);
        float cn = fg * c32[m * H_ + col] + ig * gt;
        float hn = og * tanhf(cn);
        c32[m * H_ + col] = cn;
        h32[m * H_ + col] = hn;
      }
    }
    __syncthreads();
  }

  // -- loss reduction + atomic accumulate (loss/T) --
  red[tid] = lacc;
  __syncthreads();
  for (int s = NTHREADS / 2; s > 0; s >>= 1) {
    if (tid < s) red[tid] += red[tid + s];
    __syncthreads();
  }
  if (tid == 0) atomicAdd(out + (size_t)B_ * T_ * F_, red[0] * (1.0f / T_));
}

// ---------------- host launcher ----------------
extern "C" void kernel_launch(void* const* d_in, const int* in_sizes, int n_in,
                              void* d_out, int out_size, void* d_ws, size_t ws_size,
                              hipStream_t stream) {
  const float* x     = (const float*)d_in[0];
  const float* m     = (const float*)d_in[1];
  const float* d     = (const float*)d_in[2];
  const float* tgt   = (const float*)d_in[3];
  const float* ev    = (const float*)d_in[4];
  const float* tdhW  = (const float*)d_in[5];
  const float* tdhB  = (const float*)d_in[6];
  const float* tdxW  = (const float*)d_in[7];
  const float* tdxB  = (const float*)d_in[8];
  const float* histW = (const float*)d_in[9];
  const float* histB = (const float*)d_in[10];
  const float* featW = (const float*)d_in[11];
  const float* featB = (const float*)d_in[12];
  const float* wcW   = (const float*)d_in[13];
  const float* wcB   = (const float*)d_in[14];
  const float* Wih   = (const float*)d_in[15];
  const float* Whh   = (const float*)d_in[16];
  const float* bih   = (const float*)d_in[17];
  const float* bhh   = (const float*)d_in[18];
  float* out = (float*)d_out;

  // workspace layout (bytes): packed f16 weight fragments, then aux f32
  char* ws = (char*)d_ws;
  _Float16* tdhP  = (_Float16*)(ws + 0);        // 32*4  frags * 1KB = 128KB
  _Float16* histP = (_Float16*)(ws + 131072);   // 8*16  frags       = 128KB
  _Float16* featP = (_Float16*)(ws + 262144);   // 8*4   frags       =  32KB
  _Float16* wcP   = (_Float16*)(ws + 294912);   // 8*8   frags       =  64KB
  _Float16* ihP   = (_Float16*)(ws + 360448);   // 128*8 frags       =   1MB
  _Float16* hhP   = (_Float16*)(ws + 1409024);  // 128*16 frags      =   2MB
  float* diag  = (float*)(ws + 3506176);        // [F]
  float* denom = (float*)(ws + 3506688);        // [T]
  float* gbias = (float*)(ws + 3507200);        // [4H]

  auto frags = [](int O, int I) { return (O / 16) * (I / 32); };
  pack_w<<<frags(512, 128),  256, 0, stream>>>(tdhW,  tdhP,  512, 128, 0);
  pack_w<<<frags(128, 512),  256, 0, stream>>>(histW, histP, 128, 512, 0);
  pack_w<<<frags(128, 128),  256, 0, stream>>>(featW, featP, 128, 128, 1); // zero diagonal
  pack_w<<<frags(128, 256),  256, 0, stream>>>(wcW,   wcP,   128, 256, 0);
  pack_w<<<frags(2048, 256), 256, 0, stream>>>(Wih,   ihP,   2048, 256, 0);
  pack_w<<<frags(2048, 512), 256, 0, stream>>>(Whh,   hhP,   2048, 512, 0);
  prep_aux<<<T_, 256, 0, stream>>>(tdxW, ev, bih, bhh, diag, denom, gbias,
                                   out + (size_t)B_ * T_ * F_);

  const size_t smem = 140288;   // ~137KB dynamic LDS, fits 320KB WGP budget
  rits_main<<<B_ / BT, NTHREADS, smem, stream>>>(
      x, m, d, tgt, ev, tdhB, tdxB, histB, featB, wcB, gbias,
      (const uint4*)tdhP, (const uint4*)histP, (const uint4*)featP,
      (const uint4*)wcP, (const uint4*)ihP, (const uint4*)hhP,
      diag, denom, out);
}
